// RWA_58394375356911
// MI455X (gfx1250) — matrix operand order repaired and checked by reading, not compiled, not run on hardware
//
#include <hip/hip_runtime.h>

// Problem constants (match reference)
#define DIN_  128
#define DOUT_ 128
#define BSZ_  256
#define LSEQ_ 2048

typedef __attribute__((ext_vector_type(16))) __bf16         v16bf;
typedef __attribute__((ext_vector_type(16))) unsigned short v16u;
typedef __attribute__((ext_vector_type(8)))  float          v8f;
typedef __attribute__((ext_vector_type(4)))  int            v4i;

union BFrag { v16u u; v16bf b; };

#if __has_builtin(__builtin_amdgcn_global_load_async_to_lds_b128)
#define HAVE_ASYNC_LDS 1
#else
#define HAVE_ASYNC_LDS 0
#endif

static __device__ __forceinline__ void wait_asynccnt0() {
#if __has_builtin(__builtin_amdgcn_s_wait_asynccnt)
    __builtin_amdgcn_s_wait_asynccnt(0);
#else
    asm volatile("s_wait_asynccnt 0x0" ::: "memory");
#endif
}

// Native HW conversions (v_cvt_*bf16*) instead of manual RNE bit-twiddling.
static __device__ __forceinline__ unsigned short f32_to_bf16(float f) {
    return __builtin_bit_cast(unsigned short, (__bf16)f);
}
static __device__ __forceinline__ float bf16_to_f32(unsigned short s) {
    return (float)__builtin_bit_cast(__bf16, s);
}

// gfx1250 has a hardware tanh (TRANS op) — one instruction vs ~20-op poly.
static __device__ __forceinline__ float fast_tanh(float x) {
#if __has_builtin(__builtin_amdgcn_tanhf)
    return __builtin_amdgcn_tanhf(x);
#elif __has_builtin(__builtin_amdgcn_tanh_f32)
    return __builtin_amdgcn_tanh_f32(x);
#else
    return tanhf(x);
#endif
}

// ---------------------------------------------------------------------------
// Phase 1: U = x@u_w^T + u_b ; Gx = x@g_wx^T + g_b ; Ax = x@a_wx^T
// Fused GEMM (B*L,128) x (128,384). 64-row M tile / workgroup, 4 waves.
// x tile staged f32 via async global->LDS (ASYNCcnt); bf16 WMMA 16x16x32,
// f32 accumulate (bias folded into C init); results stored bf16.
// ---------------------------------------------------------------------------
__global__ __launch_bounds__(128) void rwa_phase1(
    const float* __restrict__ x,
    const float* __restrict__ u_w, const float* __restrict__ u_b,
    const float* __restrict__ g_w, const float* __restrict__ g_b,
    const float* __restrict__ a_w,
    unsigned short* __restrict__ U, unsigned short* __restrict__ Gx,
    unsigned short* __restrict__ Ax)
{
    __shared__ float          lds_xf[64 * 128];       // 32 KB f32 x tile
    __shared__ unsigned short lds_w [8 * 4 * 512];    // 32 KB bf16 W chunk (8 n-tiles)

    const int tid  = threadIdx.x;
    const int wave = tid >> 5;
    const int lane = tid & 31;
    const int j    = lane & 15;
    const int hi   = lane >> 4;
    const long long row0 = (long long)blockIdx.x * 64;

    // ---- Stage x tile (64 x 128 f32 = 32 KB) via async DMA to LDS ----
#if HAVE_ASYNC_LDS
#pragma unroll
    for (int it = 0; it < 16; ++it) {
        const int idx = it * 512 + tid * 4;           // 4 floats / lane / issue
        __builtin_amdgcn_global_load_async_to_lds_b128(
            (v4i*)(x + row0 * 128 + idx),
            (v4i*)(&lds_xf[idx]), 0, 0);
    }
#else
    for (int i = tid; i < 64 * 128; i += 128)
        lds_xf[i] = x[row0 * 128 + i];
#endif

    // ---- Overlap: stage group-0 weights while the x DMA is in flight ----
    {
        const float* wsrc = u_w;
        for (int i = tid; i < 8 * 4 * 512; i += 128) {
            int frag = i >> 9, ntl = frag >> 2, ks = frag & 3;
            int w = i & 511, ln = w >> 4, e = w & 15;
            lds_w[i] = f32_to_bf16(wsrc[(ntl * 16 + (ln & 15)) * 128 + ks * 32 + (ln >> 4) * 16 + e]);
        }
    }
#if HAVE_ASYNC_LDS
    wait_asynccnt0();
#endif
    __syncthreads();

    // A fragments for this wave's 16 rows (convert f32 LDS -> bf16 once).
    v16bf afrag[4];
#pragma unroll
    for (int ks = 0; ks < 4; ++ks) {
        const float* src = &lds_xf[(wave * 16 + j) * 128 + ks * 32 + hi * 16];
        BFrag f;
#pragma unroll
        for (int e = 0; e < 16; ++e) f.u[e] = f32_to_bf16(src[e]);
        afrag[ks] = f.b;
    }

    for (int grp = 0; grp < 3; ++grp) {
        const float*    bias = (grp == 0) ? u_b : (grp == 1) ? g_b : nullptr;
        unsigned short* dst  = (grp == 0) ? U   : (grp == 1) ? Gx  : Ax;

        if (grp > 0) {
            const float* wsrc = (grp == 1) ? g_w : a_w;   // rows are DIN+DOUT wide
            __syncthreads();
            for (int i = tid; i < 8 * 4 * 512; i += 128) {
                int frag = i >> 9, ntl = frag >> 2, ks = frag & 3;
                int w = i & 511, ln = w >> 4, e = w & 15;
                lds_w[i] = f32_to_bf16(wsrc[(ntl * 16 + (ln & 15)) * 256 + ks * 32 + (ln >> 4) * 16 + e]);
            }
            __syncthreads();
        }

#pragma unroll
        for (int ntl = 0; ntl < 8; ++ntl) {
            const int col = ntl * 16 + j;
            const float bv = bias ? bias[col] : 0.0f;
            v8f c;
#pragma unroll
            for (int r = 0; r < 8; ++r) c[r] = bv;
#pragma unroll
            for (int ks = 0; ks < 4; ++ks) {
                v16bf bfrag = *reinterpret_cast<const v16bf*>(
                    &lds_w[(ntl * 4 + ks) * 512 + lane * 16]);
                c = __builtin_amdgcn_wmma_f32_16x16x32_bf16(
                        false, afrag[ks], false, bfrag, (short)0, c, false, false);
            }
            const long long rbase = row0 + wave * 16 + hi * 8;
#pragma unroll
            for (int r = 0; r < 8; ++r)
                dst[(rbase + r) * 128 + col] = f32_to_bf16(c[r]);
        }
    }
}

// ---------------------------------------------------------------------------
// Phase 2: sequential scan over L. One workgroup per 16-row batch tile.
// 8 waves; wave w owns output n-tile w. Recurrent weights g_wh/a_wh are
// register-resident bf16 B-fragments; h is double-buffered bf16 in LDS
// (one barrier per step); U/Gx/Ax loads software-pipelined one step ahead.
// ---------------------------------------------------------------------------
__global__ __launch_bounds__(256) void rwa_phase2(
    const float* __restrict__ s0,
    const float* __restrict__ g_w,
    const float* __restrict__ a_w,
    const unsigned short* __restrict__ U,
    const unsigned short* __restrict__ Gx,
    const unsigned short* __restrict__ Ax,
    float* __restrict__ out)
{
    __shared__ unsigned short lds_h[2][16 * 128];   // 2 x 4 KB bf16 hidden state

    const int tid  = threadIdx.x;
    const int wave = tid >> 5;           // n-tile index 0..7
    const int lane = tid & 31;
    const int j    = lane & 15;
    const int hi   = lane >> 4;
    const int b0   = blockIdx.x * 16;
    const int col  = wave * 16 + j;

    // Recurrent-weight B fragments, resident in VGPRs for all 2048 steps:
    // g_wh = g_w[:,128:], a_wh = a_w[:,128:].
    BFrag bg[4], ba[4];
#pragma unroll
    for (int ks = 0; ks < 4; ++ks) {
        const int k = ks * 32 + hi * 16;
#pragma unroll
        for (int e = 0; e < 16; ++e) {
            const int o = wave * 16 + j;
            bg[ks].u[e] = f32_to_bf16(g_w[o * 256 + 128 + k + e]);
            ba[ks].u[e] = f32_to_bf16(a_w[o * 256 + 128 + k + e]);
        }
    }

    // h0 = tanh(s0) broadcast over the batch tile (buffer 0).
    for (int i = tid; i < 16 * 128; i += 256)
        lds_h[0][i] = f32_to_bf16(fast_tanh(s0[i & 127]));
    __syncthreads();

    float num[8], den[8], amax[8];
#pragma unroll
    for (int r = 0; r < 8; ++r) { num[r] = 0.f; den[r] = 0.f; amax[r] = 1e-38f; }

    size_t base[8];
#pragma unroll
    for (int r = 0; r < 8; ++r)
        base[r] = (size_t)(b0 + r + 8 * hi) * (size_t)LSEQ_ * 128 + (size_t)col;

    // Prime the pipeline with step-0 operands.
    unsigned short cu[8], cg[8], ca[8];
#pragma unroll
    for (int r = 0; r < 8; ++r) {
        cg[r] = Gx[base[r]];
        ca[r] = Ax[base[r]];
        cu[r] = U [base[r]];
    }

    for (int t = 0; t < LSEQ_; ++t) {
        // Issue next-step loads first: their latency hides under the WMMA
        // chain + tanh/exp below (clamped index avoids a branch / OOB).
        const size_t noff = (size_t)((t + 1 < LSEQ_) ? t + 1 : t) * 128;
        unsigned short nu[8], ng[8], na[8];
#pragma unroll
        for (int r = 0; r < 8; ++r) {
            ng[r] = Gx[base[r] + noff];
            na[r] = Ax[base[r] + noff];
            nu[r] = U [base[r] + noff];
        }
        if (t + 16 < LSEQ_) {   // stream 16 steps ahead into L2
            __builtin_prefetch(&Gx[base[0] + (size_t)(t + 16) * 128], 0, 1);
            __builtin_prefetch(&Ax[base[0] + (size_t)(t + 16) * 128], 0, 1);
            __builtin_prefetch(&U [base[0] + (size_t)(t + 16) * 128], 0, 1);
        }

        // A fragments: full current h row per lane (double buffer t&1).
        const unsigned short* hbuf = lds_h[t & 1];
        v16bf ah[4];
#pragma unroll
        for (int ks = 0; ks < 4; ++ks)
            ah[ks] = *reinterpret_cast<const v16bf*>(
                &hbuf[j * 128 + ks * 32 + hi * 16]);

        v8f accg, acca;
#pragma unroll
        for (int r = 0; r < 8; ++r) {
            accg[r] = bf16_to_f32(cg[r]);
            acca[r] = bf16_to_f32(ca[r]);
        }
#pragma unroll
        for (int ks = 0; ks < 4; ++ks)
            accg = __builtin_amdgcn_wmma_f32_16x16x32_bf16(
                       false, ah[ks], false, bg[ks].b, (short)0, accg, false, false);
#pragma unroll
        for (int ks = 0; ks < 4; ++ks)
            acca = __builtin_amdgcn_wmma_f32_16x16x32_bf16(
                       false, ah[ks], false, ba[ks].b, (short)0, acca, false, false);

        const size_t toff = (size_t)t * 128;
        unsigned short* hnext = lds_h[(t + 1) & 1];
#pragma unroll
        for (int r = 0; r < 8; ++r) {
            float g  = fast_tanh(accg[r]);
            float z  = bf16_to_f32(cu[r]) * g;
            float am = fmaxf(acca[r], amax[r]);
            float e  = __expf(am - amax[r]);
            num[r] += z * e;
            den[r] += e;
            amax[r] = am;
            float h = fast_tanh(num[r] / den[r]);
            __builtin_nontemporal_store(h, &out[base[r] + toff]);
            hnext[(r + 8 * hi) * 128 + col] = f32_to_bf16(h);
        }
        __syncthreads();   // h(t+1) visible; also guards WAR on hbuf reuse

        // Retire pipeline: next-step operands become current.
#pragma unroll
        for (int r = 0; r < 8; ++r) { cg[r] = ng[r]; ca[r] = na[r]; cu[r] = nu[r]; }
    }
}

// ---------------------------------------------------------------------------
extern "C" void kernel_launch(void* const* d_in, const int* in_sizes, int n_in,
                              void* d_out, int out_size, void* d_ws, size_t ws_size,
                              hipStream_t stream)
{
    const float* x   = (const float*)d_in[0];
    const float* s0  = (const float*)d_in[1];
    const float* u_w = (const float*)d_in[2];
    const float* u_b = (const float*)d_in[3];
    const float* g_w = (const float*)d_in[4];
    const float* g_b = (const float*)d_in[5];
    const float* a_w = (const float*)d_in[6];
    float* out = (float*)d_out;

    const size_t BL = (size_t)BSZ_ * (size_t)LSEQ_;
    // Workspace: U, Gx, Ax staging as bf16 (3 * 128 MB).
    unsigned short* U  = (unsigned short*)d_ws;
    unsigned short* Gx = U  + BL * 128;
    unsigned short* Ax = Gx + BL * 128;

    rwa_phase1<<<(int)(BL / 64), 128, 0, stream>>>(x, u_w, u_b, g_w, g_b, a_w, U, Gx, Ax);
    rwa_phase2<<<BSZ_ / 16, 256, 0, stream>>>(s0, g_w, a_w, U, Gx, Ax, out);
}
